// CustomSTFT_32409823216133
// MI455X (gfx1250) — matrix operand order repaired and checked by reading, not compiled, not run on hardware
//
#include <hip/hip_runtime.h>
#include <hip/hip_bf16.h>

// ---------------------------------------------------------------------------
// STFT -> mag/phase -> iSTFT round trip for MI455X (gfx1250, wave32, WMMA).
//
// GEMMs run as split-bf16 WMMA (hi/lo decomposition, 3x
// v_wmma_f32_16x16x32_bf16 per K=32 chunk, f32 accumulators) for ~f32
// accuracy at bf16 matrix-core rates. All f32->bf16 splitting is hoisted
// into prep kernels / epilogues; operands are pre-tiled into the exact
// WMMA fragment layout so the hot loops are global_load_b128 + v_wmma.
// ---------------------------------------------------------------------------

typedef __attribute__((ext_vector_type(16))) __bf16 v16bf;
typedef __attribute__((ext_vector_type(8)))  __bf16 v8bf;
typedef __attribute__((ext_vector_type(8)))  float  v8f;

#define B_BATCH   16
#define T_LEN     240000
#define N_FFT     800
#define HOP       200
#define PADW      400            // N_FFT/2 edge padding
#define XP_LEN    240816         // 240000 + 2*PADW + 16 tail pad (b128 safety)
#define F_FRAMES  1201
#define F_PAD     1216           // 76 * 16
#define FT_TILES  76
#define K_FREQ    401
#define K_PAD     416            // 26 * 16
#define KT_TILES  26
#define NT_TILES  50             // 800 / 16
#define NC_FWD    25             // 800 / 32 K-chunks (forward)
#define NC_BWD    13             // 416 / 32 K-chunks (backward)

__device__ __forceinline__ int clampi(int v, int lo, int hi) {
  return v < lo ? lo : (v > hi ? hi : v);
}

__device__ __forceinline__ v16bf cat8(v8bf lo, v8bf hi) {
  return __builtin_shufflevector(lo, hi, 0, 1, 2, 3, 4, 5, 6, 7,
                                         8, 9, 10, 11, 12, 13, 14, 15);
}

__device__ __forceinline__ v8bf ldb128(const __bf16* p) {
  return *(const v8bf*)p;   // 16B aligned by construction -> global_load_b128
}

__device__ __forceinline__ v8f wmma_bf16(v16bf a, v16bf b, v8f c) {
  // (neg_a, A, neg_b, B, c_mod, C, reuse_a, reuse_b)
  return __builtin_amdgcn_wmma_f32_16x16x32_bf16(false, a, false, b,
                                                 (short)0, c, false, false);
}

// ---------------------------------------------------------------------------
// Kernel 0: zero the overlap-add destination (harness poisons d_out).
// ---------------------------------------------------------------------------
__global__ void stft_zero_out(float* __restrict__ p, long n) {
  long i = (long)blockIdx.x * blockDim.x + threadIdx.x;
  if (i < n) p[i] = 0.0f;
}

// ---------------------------------------------------------------------------
// Prep A: edge-padded signal, split into bf16 hi/lo planes.
//   xp[b][j] = x[b][clamp(j - PADW, 0, T-1)],  j in [0, XP_LEN)
// ---------------------------------------------------------------------------
__global__ void stft_prep_x(const float* __restrict__ x,
                            __bf16* __restrict__ XPH,
                            __bf16* __restrict__ XPL) {
  long i = (long)blockIdx.x * blockDim.x + threadIdx.x;
  if (i >= (long)B_BATCH * XP_LEN) return;
  int b = (int)(i / XP_LEN);
  int j = (int)(i % XP_LEN);
  float v = x[(long)b * T_LEN + clampi(j - PADW, 0, T_LEN - 1)];
  __bf16 h = (__bf16)v;
  XPH[i] = h;
  XPL[i] = (__bf16)(v - (float)h);
}

// ---------------------------------------------------------------------------
// Prep B (forward): pre-tile DFT weights into WMMA B-fragment layout.
// Fragment layout: [ktile][chunk][lane][e0..15], element e <-> inner-K
//   K = chunk*32 + (e>>3)*16 + (lane>>4)*8 + (e&7) ; column (freq) kq = ktile*16 + (lane&15)
// ---------------------------------------------------------------------------
__global__ void stft_prep_wf(const float* __restrict__ wr,
                             const float* __restrict__ wi,
                             __bf16* __restrict__ RH, __bf16* __restrict__ RL,
                             __bf16* __restrict__ IH, __bf16* __restrict__ IL) {
  long i = (long)blockIdx.x * blockDim.x + threadIdx.x;
  if (i >= (long)KT_TILES * NC_FWD * 32 * 16) return;
  int e    = (int)(i & 15);
  int lane = (int)((i >> 4) & 31);
  int c    = (int)((i >> 9) % NC_FWD);
  int kt   = (int)(i / (NC_FWD * 512));
  int kq   = clampi(kt * 16 + (lane & 15), 0, K_FREQ - 1);
  int K    = c * 32 + (e >> 3) * 16 + (lane >> 4) * 8 + (e & 7);  // < 800
  float r = wr[(long)kq * N_FFT + K];
  float m = wi[(long)kq * N_FFT + K];
  __bf16 rh = (__bf16)r, ih = (__bf16)m;
  RH[i] = rh; RL[i] = (__bf16)(r - (float)rh);
  IH[i] = ih; IL[i] = (__bf16)(m - (float)ih);
}

// ---------------------------------------------------------------------------
// Prep B (backward): same fragment layout over [ntile][chunk][lane][e].
//   inner-K (freq) kr = chunk*32 + (e>>3)*16 + (lane>>4)*8 + (e&7) (clamped; A is 0 there)
//   column n = ntile*16 + (lane&15)
// ---------------------------------------------------------------------------
__global__ void stft_prep_wb(const float* __restrict__ wbr,
                             const float* __restrict__ wbi,
                             __bf16* __restrict__ RH, __bf16* __restrict__ RL,
                             __bf16* __restrict__ IH, __bf16* __restrict__ IL) {
  long i = (long)blockIdx.x * blockDim.x + threadIdx.x;
  if (i >= (long)NT_TILES * NC_BWD * 32 * 16) return;
  int e    = (int)(i & 15);
  int lane = (int)((i >> 4) & 31);
  int c    = (int)((i >> 9) % NC_BWD);
  int nt   = (int)(i / (NC_BWD * 512));
  int nq   = nt * 16 + (lane & 15);                  // < 800
  int kr   = clampi(c * 32 + (e >> 3) * 16 + (lane >> 4) * 8 + (e & 7), 0, K_FREQ - 1);
  float r = wbr[(long)kr * N_FFT + nq];
  float m = wbi[(long)kr * N_FFT + nq];
  __bf16 rh = (__bf16)r, ih = (__bf16)m;
  RH[i] = rh; RL[i] = (__bf16)(r - (float)rh);
  IH[i] = ih; IL[i] = (__bf16)(m - (float)ih);
}

// ---------------------------------------------------------------------------
// Kernel 1: forward STFT (one wave per 16x16 tile of [frame, freq]).
// Hot loop per K=32 chunk: 12x global_load_b128 + 6x v_wmma. Epilogue does
// the mag/phase round trip and stores split-bf16 zero-padded spectrum planes.
// ---------------------------------------------------------------------------
__global__ __launch_bounds__(256)
void stft_fwd_wmma(const __bf16* __restrict__ XPH, const __bf16* __restrict__ XPL,
                   const __bf16* __restrict__ WRH, const __bf16* __restrict__ WRL,
                   const __bf16* __restrict__ WIH, const __bf16* __restrict__ WIL,
                   __bf16* __restrict__ RPH, __bf16* __restrict__ RPL,
                   __bf16* __restrict__ IPH, __bf16* __restrict__ IPL)
{
  const int lane = threadIdx.x & 31;
  const int wid  = (blockIdx.x << 3) + (threadIdx.x >> 5);

  int kt = wid % KT_TILES;
  int t1 = wid / KT_TILES;
  int ft = t1 % FT_TILES;
  int b  = t1 / FT_TILES;
  if (b >= B_BATCH) return;

  const int lm = lane & 15;
  const int hs = lane >> 4;

  // A fragment: lane's frame row, two contiguous 8-elt runs at K base+0, +16.
  // fc*HOP is a multiple of 8 -> 16B-aligned b128 loads.
  const int  fc = clampi(ft * 16 + lm, 0, F_FRAMES - 1);
  long ja = (long)b * XP_LEN + (long)fc * HOP + hs * 8;   // chunk 0 base
  // B fragment base: [kt][c][lane][16]
  long jb = ((long)kt * NC_FWD * 32 + lane) * 16;

  v8f accr = {0.f, 0.f, 0.f, 0.f, 0.f, 0.f, 0.f, 0.f};
  v8f acci = accr;

  for (int c = 0; c < NC_FWD; ++c, ja += 32, jb += 512) {
    v16bf ah = cat8(ldb128(XPH + ja), ldb128(XPH + ja + 16));
    v16bf al = cat8(ldb128(XPL + ja), ldb128(XPL + ja + 16));
    v16bf rh = cat8(ldb128(WRH + jb), ldb128(WRH + jb + 8));
    v16bf rl = cat8(ldb128(WRL + jb), ldb128(WRL + jb + 8));
    v16bf ih = cat8(ldb128(WIH + jb), ldb128(WIH + jb + 8));
    v16bf il = cat8(ldb128(WIL + jb), ldb128(WIL + jb + 8));
    // split-bf16 f32 emulation: Ah*Bh + Ah*Bl + Al*Bh
    accr = wmma_bf16(ah, rh, accr);
    accr = wmma_bf16(ah, rl, accr);
    accr = wmma_bf16(al, rh, accr);
    acci = wmma_bf16(ah, ih, acci);
    acci = wmma_bf16(ah, il, acci);
    acci = wmma_bf16(al, ih, acci);
  }

  // Epilogue: mag/phase reconstruction (== mag*cos/sin(atan2), incl. the
  // i==0 && r<0 case), then split-bf16 store of zero-padded planes.
#pragma unroll
  for (int v = 0; v < 8; ++v) {
    const int fo = ft * 16 + v + hs * 8;   // C row (frame)
    const int ko = kt * 16 + lm;           // C col (freq)
    float r = accr[v], i = acci[v];
    float h2  = r * r + i * i;
    float mag = sqrtf(h2 + 1e-14f);
    float s   = (h2 > 0.0f) ? mag * rsqrtf(h2) : 0.0f;
    float rp  = (h2 > 0.0f) ? r * s : mag;  // atan2(0,0)=0 -> cos=1
    float ip  = i * s;
    const bool valid = (fo < F_FRAMES) && (ko < K_FREQ);
    rp = valid ? rp : 0.0f;
    ip = valid ? ip : 0.0f;
    __bf16 rh = (__bf16)rp, ih2 = (__bf16)ip;
    const long o = ((long)b * F_PAD + fo) * K_PAD + ko;
    RPH[o] = rh; RPL[o] = (__bf16)(rp - (float)rh);
    IPH[o] = ih2; IPL[o] = (__bf16)(ip - (float)ih2);
  }
}

// ---------------------------------------------------------------------------
// Kernel 2: inverse STFT + overlap-add (one wave per 16x16 [frame, n] tile).
// Hot loop per K=32 chunk: 16x global_load_b128 + 6x v_wmma.
// ---------------------------------------------------------------------------
__global__ __launch_bounds__(256)
void stft_bwd_wmma(const __bf16* __restrict__ RPH, const __bf16* __restrict__ RPL,
                   const __bf16* __restrict__ IPH, const __bf16* __restrict__ IPL,
                   const __bf16* __restrict__ WRH, const __bf16* __restrict__ WRL,
                   const __bf16* __restrict__ WIH, const __bf16* __restrict__ WIL,
                   float* __restrict__ out)
{
  const int lane = threadIdx.x & 31;
  const int wid  = (blockIdx.x << 3) + (threadIdx.x >> 5);

  int nt = wid % NT_TILES;
  int t1 = wid / NT_TILES;
  int ft = t1 % FT_TILES;
  int b  = t1 / FT_TILES;
  if (b >= B_BATCH) return;

  const int lm = lane & 15;
  const int hs = lane >> 4;

  // A: spectrum row (K_PAD-contiguous, zero-padded) -> aligned b128 runs.
  long ja = ((long)b * F_PAD + ft * 16 + lm) * K_PAD + hs * 8;
  // B: pre-tiled inverse weights [nt][c][lane][16]
  long jb = ((long)nt * NC_BWD * 32 + lane) * 16;

  v8f accr = {0.f, 0.f, 0.f, 0.f, 0.f, 0.f, 0.f, 0.f};
  v8f acci = accr;

  for (int c = 0; c < NC_BWD; ++c, ja += 32, jb += 512) {
    v16bf arh = cat8(ldb128(RPH + ja), ldb128(RPH + ja + 16));
    v16bf arl = cat8(ldb128(RPL + ja), ldb128(RPL + ja + 16));
    v16bf aih = cat8(ldb128(IPH + ja), ldb128(IPH + ja + 16));
    v16bf ail = cat8(ldb128(IPL + ja), ldb128(IPL + ja + 16));
    v16bf brh = cat8(ldb128(WRH + jb), ldb128(WRH + jb + 8));
    v16bf brl = cat8(ldb128(WRL + jb), ldb128(WRL + jb + 8));
    v16bf bih = cat8(ldb128(WIH + jb), ldb128(WIH + jb + 8));
    v16bf bil = cat8(ldb128(WIL + jb), ldb128(WIL + jb + 8));
    accr = wmma_bf16(arh, brh, accr);
    accr = wmma_bf16(arh, brl, accr);
    accr = wmma_bf16(arl, brh, accr);
    acci = wmma_bf16(aih, bih, acci);
    acci = wmma_bf16(aih, bil, acci);
    acci = wmma_bf16(ail, bih, acci);
  }

  // Overlap-add scatter (4 adds per output sample) via global_atomic_add_f32.
#pragma unroll
  for (int v = 0; v < 8; ++v) {
    const int f = ft * 16 + v + hs * 8;
    const int n = nt * 16 + lm;
    if (f < F_FRAMES) {
      const int t = f * HOP + n - PADW;
      if ((unsigned)t < (unsigned)T_LEN) {
        atomicAdd(out + (long)b * T_LEN + t, accr[v] - acci[v]);
      }
    }
  }
}

// ---------------------------------------------------------------------------
// Launcher. Inputs: x, w_fwd_r, w_fwd_i, w_bwd_r, w_bwd_i (all f32).
// Workspace (~86 MB): split signal + pre-tiled weights + split spectrum.
// ---------------------------------------------------------------------------
extern "C" void kernel_launch(void* const* d_in, const int* in_sizes, int n_in,
                              void* d_out, int out_size, void* d_ws, size_t ws_size,
                              hipStream_t stream) {
  const float* x   = (const float*)d_in[0];
  const float* wfr = (const float*)d_in[1];
  const float* wfi = (const float*)d_in[2];
  const float* wbr = (const float*)d_in[3];
  const float* wbi = (const float*)d_in[4];
  float* out = (float*)d_out;

  // carve workspace (all section sizes are multiples of 8 elems -> 16B aligned)
  __bf16* p = (__bf16*)d_ws;
  const long SZ_XP = (long)B_BATCH * XP_LEN;              // 3,853,056
  const long SZ_WF = (long)KT_TILES * NC_FWD * 32 * 16;   //   332,800
  const long SZ_WB = (long)NT_TILES * NC_BWD * 32 * 16;   //   332,800
  const long SZ_SP = (long)B_BATCH * F_PAD * K_PAD;       // 8,093,696
  __bf16 *XPH = p;            p += SZ_XP;
  __bf16 *XPL = p;            p += SZ_XP;
  __bf16 *WFRH = p;           p += SZ_WF;
  __bf16 *WFRL = p;           p += SZ_WF;
  __bf16 *WFIH = p;           p += SZ_WF;
  __bf16 *WFIL = p;           p += SZ_WF;
  __bf16 *WBRH = p;           p += SZ_WB;
  __bf16 *WBRL = p;           p += SZ_WB;
  __bf16 *WBIH = p;           p += SZ_WB;
  __bf16 *WBIL = p;           p += SZ_WB;
  __bf16 *RPH = p;            p += SZ_SP;
  __bf16 *RPL = p;            p += SZ_SP;
  __bf16 *IPH = p;            p += SZ_SP;
  __bf16 *IPL = p;            p += SZ_SP;

  const long nout = (long)B_BATCH * T_LEN;                // 3,840,000
  stft_zero_out<<<(int)((nout + 255) / 256), 256, 0, stream>>>(out, nout);
  stft_prep_x<<<(int)((SZ_XP + 255) / 256), 256, 0, stream>>>(x, XPH, XPL);
  stft_prep_wf<<<(int)((SZ_WF + 255) / 256), 256, 0, stream>>>(
      wfr, wfi, WFRH, WFRL, WFIH, WFIL);
  stft_prep_wb<<<(int)((SZ_WB + 255) / 256), 256, 0, stream>>>(
      wbr, wbi, WBRH, WBRL, WBIH, WBIL);

  // one wave per 16x16 tile, 8 waves per block (wave32)
  const int fwd_blocks = (B_BATCH * FT_TILES * KT_TILES) / 8;   // 3952
  stft_fwd_wmma<<<fwd_blocks, 256, 0, stream>>>(
      XPH, XPL, WFRH, WFRL, WFIH, WFIL, RPH, RPL, IPH, IPL);

  const int bwd_blocks = (B_BATCH * FT_TILES * NT_TILES) / 8;   // 7600
  stft_bwd_wmma<<<bwd_blocks, 256, 0, stream>>>(
      RPH, RPL, IPH, IPL, WBRH, WBRL, WBIH, WBIL, out);
}
